// ScaledDotProductAttention_86217173500224
// MI455X (gfx1250) — matrix operand order repaired and checked by reading, not compiled, not run on hardware
//
#include <hip/hip_runtime.h>
#include <hip/hip_bf16.h>
#include <math.h>

// ---------------------------------------------------------------------------
// Scaled dot-product attention (QKV projection + flash attention) for
// MI455X / gfx1250 (wave32, WMMA 16x16x32 f16->f32, async global->LDS).
//
// B=2, H=16, S=2048, D=64.
// Workspace layout (f16): Q [B*H*S, D], K [B*H*S, D], Vt [B*H, D, S]
//   -> 3 * 8 MiB = 24 MiB of d_ws.
// ---------------------------------------------------------------------------

typedef __attribute__((ext_vector_type(16))) _Float16 v16h;
typedef __attribute__((ext_vector_type(8)))  float    v8f;
typedef __attribute__((ext_vector_type(4)))  unsigned int u32x4;

union FragH {
    v16h     v;
    _Float16 h[16];
    u32x4    q[2];
};

constexpr int   kBatch = 2, kHeads = 16, kSeq = 2048, kDim = 64;
constexpr float kLog2e = 1.44269504088896f;
constexpr int   kLdsStride = 72;     // f16 elements per LDS row (64 + 8 pad)

#define WMMA_F32_F16(A, Bf, C) \
    __builtin_amdgcn_wmma_f32_16x16x32_f16(false, (A), false, (Bf), (short)0, (C), false, false)

// ---------------------------------------------------------------------------
// Kernel 1: y = x @ W^T + b  for W in {Wq, Wk, Wv}.  One wave per 16-row tile.
// Q gets the 1/sqrt(D) softmax scale folded in.  V is stored transposed.
// ---------------------------------------------------------------------------
__global__ __launch_bounds__(32) void qkv_proj_kernel(
    const float* __restrict__ x,
    const float* __restrict__ Wq, const float* __restrict__ bq,
    const float* __restrict__ Wk, const float* __restrict__ bk,
    const float* __restrict__ Wv, const float* __restrict__ bv,
    _Float16* __restrict__ qbuf, _Float16* __restrict__ kbuf,
    _Float16* __restrict__ vtbuf)
{
    const int lane = threadIdx.x;
    const int tile = blockIdx.x;   // 16-row tile over flattened B*H*S rows
    const int proj = blockIdx.y;   // 0 = Q, 1 = K, 2 = V

    const float* W   = (proj == 0) ? Wq : (proj == 1) ? Wk : Wv;
    const float* bia = (proj == 0) ? bq : (proj == 1) ? bk : bv;

    const int  m      = lane & 15;
    const int  aklo   = (lane < 16) ? 0 : 8;    // A-fragment K sub-block
    const int  n      = lane & 15;
    const int  bkhi   = (lane < 16) ? 0 : 8;    // B-fragment K sub-block
    const int  rowoff = (lane < 16) ? 0 : 8;    // C-layout row half
    const long row0   = (long)tile * 16;

    // ---- A fragments: 16x64 slab of x, fp32 -> f16 -------------------------
    FragH a[2];
    {
        const float* xr = x + (row0 + m) * (long)kDim;
        #pragma unroll
        for (int kk = 0; kk < 2; ++kk) {
            const int base = kk * 32 + aklo;
            float4 f0 = *(const float4*)(xr + base);
            float4 f1 = *(const float4*)(xr + base + 4);
            float4 f2 = *(const float4*)(xr + base + 16);
            float4 f3 = *(const float4*)(xr + base + 20);
            a[kk].h[0]  = (_Float16)f0.x; a[kk].h[1]  = (_Float16)f0.y;
            a[kk].h[2]  = (_Float16)f0.z; a[kk].h[3]  = (_Float16)f0.w;
            a[kk].h[4]  = (_Float16)f1.x; a[kk].h[5]  = (_Float16)f1.y;
            a[kk].h[6]  = (_Float16)f1.z; a[kk].h[7]  = (_Float16)f1.w;
            a[kk].h[8]  = (_Float16)f2.x; a[kk].h[9]  = (_Float16)f2.y;
            a[kk].h[10] = (_Float16)f2.z; a[kk].h[11] = (_Float16)f2.w;
            a[kk].h[12] = (_Float16)f3.x; a[kk].h[13] = (_Float16)f3.y;
            a[kk].h[14] = (_Float16)f3.z; a[kk].h[15] = (_Float16)f3.w;
        }
    }

    // ---- 4 N-tiles x 2 K-chunks of WMMA ------------------------------------
    #pragma unroll
    for (int nt = 0; nt < 4; ++nt) {
        v8f c = {};
        #pragma unroll
        for (int kk = 0; kk < 2; ++kk) {
            FragH bf;
            // B[k][n] = W[nt*16+n][k]  (y = x @ W^T, W stored [out,in])
            const float* wr = W + (long)(nt * 16 + n) * kDim + kk * 32 + bkhi;
            float4 f0 = *(const float4*)(wr);
            float4 f1 = *(const float4*)(wr + 4);
            float4 f2 = *(const float4*)(wr + 16);
            float4 f3 = *(const float4*)(wr + 20);
            bf.h[0]  = (_Float16)f0.x; bf.h[1]  = (_Float16)f0.y;
            bf.h[2]  = (_Float16)f0.z; bf.h[3]  = (_Float16)f0.w;
            bf.h[4]  = (_Float16)f1.x; bf.h[5]  = (_Float16)f1.y;
            bf.h[6]  = (_Float16)f1.z; bf.h[7]  = (_Float16)f1.w;
            bf.h[8]  = (_Float16)f2.x; bf.h[9]  = (_Float16)f2.y;
            bf.h[10] = (_Float16)f2.z; bf.h[11] = (_Float16)f2.w;
            bf.h[12] = (_Float16)f3.x; bf.h[13] = (_Float16)f3.y;
            bf.h[14] = (_Float16)f3.z; bf.h[15] = (_Float16)f3.w;
            c = WMMA_F32_F16(a[kk].v, bf.v, c);
        }
        const float bval = bia[nt * 16 + n];
        #pragma unroll
        for (int r = 0; r < 8; ++r) c[r] += bval;
        if (proj == 0) {            // fold softmax scale 1/sqrt(64) into Q
            #pragma unroll
            for (int r = 0; r < 8; ++r) c[r] *= 0.125f;
        }

        if (proj == 2) {
            // Transposed V store: Vt[bh][d][s]; per-lane 8 consecutive s -> b128
            const long bh = row0 / kSeq;
            const long s0 = row0 % kSeq;
            _Float16* vp = vtbuf + (bh * kDim + (nt * 16 + n)) * (long)kSeq + s0 + rowoff;
            FragH pk;
            #pragma unroll
            for (int r = 0; r < 8; ++r) pk.h[r] = (_Float16)c[r];
            *(u32x4*)vp = pk.q[0];
        } else {
            _Float16* obuf = (proj == 0) ? qbuf : kbuf;
            #pragma unroll
            for (int r = 0; r < 8; ++r)
                obuf[(row0 + rowoff + r) * kDim + nt * 16 + n] = (_Float16)c[r];
        }
    }
}

// ---------------------------------------------------------------------------
// Cooperative async stage of a 64-row x 128-byte tile into LDS (row pad 144B).
// 128 threads -> 4 chunks of 16B per thread -> 4 async instructions per wave.
// ---------------------------------------------------------------------------
__device__ __forceinline__ void async_stage_tile(const char* gbase, long rowStrideB,
                                                 unsigned ldsBase, int tid)
{
    #pragma unroll
    for (int c = 0; c < 4; ++c) {
        const int o   = c * 2048 + tid * 16;   // linear offset in packed tile
        const int row = o >> 7;
        const int col = o & 127;
        const char*    g = gbase + (long)row * rowStrideB + col;
        const unsigned l = ldsBase + (unsigned)(row * (kLdsStride * 2) + col);
        asm volatile("global_load_async_to_lds_b128 %0, %1, off"
                     :: "v"(l), "v"(g) : "memory");
    }
}

// ---------------------------------------------------------------------------
// Kernel 2: flash attention.  4 waves/block; each wave owns a 16-row q-tile.
// K/V blocks double-buffered in LDS via async global->LDS DMA (ASYNCcnt).
// ---------------------------------------------------------------------------
__global__ __launch_bounds__(128) void flash_attn_kernel(
    const _Float16* __restrict__ qbuf, const _Float16* __restrict__ kbuf,
    const _Float16* __restrict__ vtbuf, float* __restrict__ out)
{
    const int tid  = threadIdx.x;
    const int lane = tid & 31;
    const int wid  = tid >> 5;
    const int nQB  = kSeq / 64;
    const int bh   = blockIdx.x / nQB;
    const int qblk = blockIdx.x % nQB;
    const int q0   = qblk * 64 + wid * 16;     // q-row base within S

    const _Float16* Qp = qbuf  + ((long)bh * kSeq + q0) * kDim;
    const _Float16* Kp = kbuf  + (long)bh * kSeq * kDim;
    const _Float16* Vp = vtbuf + (long)bh * kDim * kSeq;   // [d][s]

    const int m      = lane & 15;
    const int aklo   = (lane < 16) ? 0 : 8;
    const int n      = lane & 15;
    const int bkhi   = (lane < 16) ? 0 : 8;
    const int rowoff = (lane < 16) ? 0 : 8;

    // ---- Q A-fragments (softmax scale already folded in) -------------------
    FragH qa[2];
    {
        const _Float16* qr = Qp + (long)m * kDim;
        #pragma unroll
        for (int kk = 0; kk < 2; ++kk) {
            qa[kk].q[0] = *(const u32x4*)(qr + kk * 32 + aklo);
            qa[kk].q[1] = *(const u32x4*)(qr + kk * 32 + aklo + 16);
        }
    }

    v8f   acc[4] = {};
    float mrow[8], lrow[8];
    #pragma unroll
    for (int r = 0; r < 8; ++r) { mrow[r] = -3.0e38f; lrow[r] = 0.0f; }

    __shared__ __align__(16) _Float16 ktile[2][64][kLdsStride];  // 18 KiB
    __shared__ __align__(16) _Float16 vtile[2][64][kLdsStride];  // 18 KiB
    __shared__ __align__(16) _Float16 pbuf[4][16][kLdsStride];   //  9 KiB

    const unsigned ldsK[2] = { (unsigned)(uintptr_t)&ktile[0][0][0],
                               (unsigned)(uintptr_t)&ktile[1][0][0] };
    const unsigned ldsV[2] = { (unsigned)(uintptr_t)&vtile[0][0][0],
                               (unsigned)(uintptr_t)&vtile[1][0][0] };

    // ---- prologue: stage k-block 0 -----------------------------------------
    async_stage_tile((const char*)Kp, (long)kDim * 2, ldsK[0], tid);
    async_stage_tile((const char*)Vp, (long)kSeq * 2, ldsV[0], tid);

    const int nblk = kSeq / 64;
    for (int i = 0; i < nblk; ++i) {
        const int cur = i & 1;
        const int kb  = i * 64;

        // stage next block into the other buffer, then wait for current block
        if (i + 1 < nblk) {
            async_stage_tile((const char*)(Kp + (long)(kb + 64) * kDim),
                             (long)kDim * 2, ldsK[1 - cur], tid);
            async_stage_tile((const char*)Vp + (long)(kb + 64) * 2,
                             (long)kSeq * 2, ldsV[1 - cur], tid);
            // 8 newer ops outstanding; in-order => block i has landed
            asm volatile("s_wait_asynccnt 0x8" ::: "memory");
        } else {
            asm volatile("s_wait_asynccnt 0x0" ::: "memory");
        }
        __syncthreads();   // everyone's async data visible

        // ---- S = Q @ K^T : 4 N-tiles x 2 K-chunks (B frags from LDS) -------
        v8f s[4];
        #pragma unroll
        for (int nt = 0; nt < 4; ++nt) {
            v8f c = {};
            #pragma unroll
            for (int kk = 0; kk < 2; ++kk) {
                FragH bf;   // B[k][n] = K[kb+nt*16+n][k]
                const _Float16* kr = &ktile[cur][nt * 16 + n][kk * 32 + bkhi];
                bf.q[0] = *(const u32x4*)(kr);
                bf.q[1] = *(const u32x4*)(kr + 16);
                c = WMMA_F32_F16(qa[kk].v, bf.v, c);
            }
            s[nt] = c;
        }

        // ---- online softmax (per C-layout row; 16-lane halves) -------------
        #pragma unroll
        for (int r = 0; r < 8; ++r) {
            float mx = fmaxf(fmaxf(s[0][r], s[1][r]), fmaxf(s[2][r], s[3][r]));
            mx = fmaxf(mx, __shfl_xor(mx, 1, 32));
            mx = fmaxf(mx, __shfl_xor(mx, 2, 32));
            mx = fmaxf(mx, __shfl_xor(mx, 4, 32));
            mx = fmaxf(mx, __shfl_xor(mx, 8, 32));
            const float newm  = fmaxf(mrow[r], mx);
            const float alpha = exp2f((mrow[r] - newm) * kLog2e);
            float rs = 0.0f;
            #pragma unroll
            for (int t = 0; t < 4; ++t) {
                const float p = exp2f((s[t][r] - newm) * kLog2e);
                s[t][r] = p;
                rs += p;
            }
            rs += __shfl_xor(rs, 1, 32);
            rs += __shfl_xor(rs, 2, 32);
            rs += __shfl_xor(rs, 4, 32);
            rs += __shfl_xor(rs, 8, 32);
            lrow[r] = lrow[r] * alpha + rs;
            mrow[r] = newm;
            #pragma unroll
            for (int t = 0; t < 4; ++t) acc[t][r] *= alpha;
        }

        // ---- P: C-layout -> LDS (f16) -> A-layout (wave-private) -----------
        #pragma unroll
        for (int t = 0; t < 4; ++t)
            #pragma unroll
            for (int r = 0; r < 8; ++r)
                pbuf[wid][rowoff + r][t * 16 + n] = (_Float16)s[t][r];

        asm volatile("s_wait_dscnt 0x0" ::: "memory");   // wave-private RAW fence

        FragH pa[2];
        #pragma unroll
        for (int kk = 0; kk < 2; ++kk) {
            const _Float16* pr = &pbuf[wid][m][kk * 32 + aklo];
            pa[kk].q[0] = *(const u32x4*)(pr);
            pa[kk].q[1] = *(const u32x4*)(pr + 16);
        }

        // ---- acc += P @ V (B frags from LDS V-transposed tile) -------------
        #pragma unroll
        for (int dt = 0; dt < 4; ++dt) {
            #pragma unroll
            for (int kk = 0; kk < 2; ++kk) {
                FragH bf;   // B[k][d] = Vt[dt*16+n][kb+k]
                const _Float16* vr = &vtile[cur][dt * 16 + n][kk * 32 + bkhi];
                bf.q[0] = *(const u32x4*)(vr);
                bf.q[1] = *(const u32x4*)(vr + 16);
                acc[dt] = WMMA_F32_F16(pa[kk].v, bf.v, acc[dt]);
            }
        }

        __syncthreads();   // all waves done reading buf[cur] before reuse
    }

    // ---- epilogue: normalize and store fp32 --------------------------------
    float* op = out + ((long)bh * kSeq + q0) * kDim;
    #pragma unroll
    for (int r = 0; r < 8; ++r) {
        const float inv = 1.0f / lrow[r];
        #pragma unroll
        for (int t = 0; t < 4; ++t)
            op[(long)(rowoff + r) * kDim + t * 16 + n] = acc[t][r] * inv;
    }
}

// ---------------------------------------------------------------------------
extern "C" void kernel_launch(void* const* d_in, const int* in_sizes, int n_in,
                              void* d_out, int out_size, void* d_ws, size_t ws_size,
                              hipStream_t stream) {
    const float* x  = (const float*)d_in[0];
    const float* Wq = (const float*)d_in[1];
    const float* bq = (const float*)d_in[2];
    const float* Wk = (const float*)d_in[3];
    const float* bk = (const float*)d_in[4];
    const float* Wv = (const float*)d_in[5];
    const float* bv = (const float*)d_in[6];
    float* out = (float*)d_out;

    const long elems = (long)kBatch * kHeads * kSeq * kDim;   // 4M f16 per buffer
    _Float16* qbuf  = (_Float16*)d_ws;
    _Float16* kbuf  = qbuf + elems;
    _Float16* vtbuf = kbuf + elems;     // needs 24 MiB of workspace total

    const int nTiles = kBatch * kHeads * kSeq / 16;           // 4096
    qkv_proj_kernel<<<dim3(nTiles, 3), 32, 0, stream>>>(
        x, Wq, bq, Wk, bk, Wv, bv, qbuf, kbuf, vtbuf);

    const int nBlocks = kBatch * kHeads * (kSeq / 64);        // 1024
    flash_attn_kernel<<<dim3(nBlocks), 128, 0, stream>>>(qbuf, kbuf, vtbuf, out);
}
